// FirDownsample2D_33303176413435
// MI455X (gfx1250) — matrix operand order repaired and checked by reading, not compiled, not run on hardware
//
#include <hip/hip_runtime.h>

typedef __attribute__((ext_vector_type(16))) __bf16 v16bf;
typedef __attribute__((ext_vector_type(8)))  float  v8f;
typedef __attribute__((ext_vector_type(4)))  unsigned v4u;
typedef __attribute__((ext_vector_type(8)))  int      v8i;
typedef __attribute__((ext_vector_type(4)))  int      v4i;

__device__ __forceinline__ unsigned short f2bf_rne(float f) {
    union { float f; unsigned u; } v; v.f = f;
    unsigned r = v.u + 0x7FFFu + ((v.u >> 16) & 1u);
    return (unsigned short)(r >> 16);
}

// ---------------------------------------------------------------------------
// Kernel 0: pack conv weights [512,512,3,3] f32 -> bf16 WMMA A-fragments.
// pk[((k)*32 + cotile)*512 + lane*16 + j],  k = kidx*16 + ciblk
// A-matrix 16x32 bf16 lane layout used here (contiguous k order; B reads match):
//   lane l: g = l>>4, m = l&15;  half j -> Klocal = 16g+j
// ---------------------------------------------------------------------------
__global__ __launch_bounds__(256) void pack_weights(const float* __restrict__ w,
                                                    unsigned short* __restrict__ pk) {
    unsigned idx = blockIdx.x * 256u + threadIdx.x;      // 9*16*32*32*16 total
    unsigned j      = idx & 15u;
    unsigned lane   = (idx >> 4) & 31u;
    unsigned cotile = (idx >> 9) & 31u;
    unsigned rest   = idx >> 14;
    unsigned ciblk  = rest & 15u;
    unsigned kidx   = rest >> 4;                         // 0..8
    if (kidx >= 9u) return;
    unsigned g = lane >> 4, m = lane & 15u;
    unsigned kl = g * 16u + j;
    unsigned co = cotile * 16u + m;
    unsigned ci = ciblk * 32u + kl;
    unsigned kh = kidx / 3u, kw = kidx % 3u;
    pk[idx] = f2bf_rne(w[((co * 512u + ci) * 3u + kh) * 3u + kw]);
}

// ---------------------------------------------------------------------------
// Kernel 1: depthwise FIR (separable (1,3,3,1)/8 per axis), pad=2.
// x: NCHW f32 [8,512,64,64]  ->  y: channels-last bf16 [8,65,65,512]
// ---------------------------------------------------------------------------
__global__ __launch_bounds__(256) void fir_nhwc(const float* __restrict__ x,
                                                unsigned short* __restrict__ y) {
    unsigned idx = blockIdx.x * 256u + threadIdx.x;      // 8*65*65*512 total
    unsigned ci   = idx & 511u;
    unsigned rest = idx >> 9;
    unsigned xx = rest % 65u; rest /= 65u;
    unsigned yy = rest % 65u;
    unsigned n  = rest / 65u;
    if (n >= 8u) return;
    const float w4[4] = {0.125f, 0.375f, 0.375f, 0.125f};
    const float* xp = x + ((n * 512u + ci) << 12);
    float acc = 0.f;
#pragma unroll
    for (int i = 0; i < 4; ++i) {
        int py = (int)yy - 2 + i;
        if ((unsigned)py >= 64u) continue;
        float rsum = 0.f;
#pragma unroll
        for (int jj = 0; jj < 4; ++jj) {
            int px = (int)xx - 2 + jj;
            if ((unsigned)px >= 64u) continue;
            rsum += w4[jj] * xp[(py << 6) + px];
        }
        acc += w4[i] * rsum;
    }
    y[idx] = f2bf_rne(acc);
}

// ---------------------------------------------------------------------------
// TDM: DMA one B tile (K-step) into LDS.
// Tile = [z=4 ho-rows][y=32 wo-cols][x=32 ci-halfs], data_size=2B.
//   dim0 stride = 2*512      = 1024 halfs  (xx step 2, NHWC C=512)
//   dim1 stride = 2*65*512   = 66560 halfs (yy step 2)
// LDS result: contiguous [n_local(128)][64B], n_local = z*32 + y.
// D# per ISA 8.3-8.6: group0 {count=1 | lds_addr | global_addr | type=2},
// group1 {data_size=1(2B), tensor_dim0/1=32, tile=32x32x4, strides}, group2 {dim2=4}.
// 6-arg builtin: (g0, g1, g2, g3, VADDR4-unused, cpol).
// ---------------------------------------------------------------------------
__device__ __forceinline__ void tdm_issue_b(unsigned lds_off, unsigned long long gaddr) {
    v4u g0; v8i g1; v4i g2; v4i g3; v8i g4;
    g0[0] = 1u;                                          // count=1 (valid user desc)
    g0[1] = lds_off;                                     // lds_addr (bytes)
    g0[2] = (unsigned)(gaddr & 0xFFFFFFFFu);             // global_addr[31:0]
    g0[3] = (unsigned)((gaddr >> 32) & 0x01FFFFFFu) | 0x80000000u;  // ga[56:32] | type=2
    g1[0] = 0x00010000;                                  // wg_mask=0, data_size=1 (2B)
    g1[1] = (int)(32u << 16);                            // tensor_dim0 = 32 (lo16)
    g1[2] = (int)(32u << 16);                            // td0 hi=0 | tensor_dim1 = 32 (lo16)
    g1[3] = (int)(32u << 16);                            // td1 hi=0 | tile_dim0 = 32
    g1[4] = (int)(32u | (4u << 16));                     // tile_dim1=32 | tile_dim2=4
    g1[5] = 1024;                                        // tensor_dim0_stride[31:0]
    g1[6] = (int)(0x0400u << 16);                        // s0[47:32]=0 | s1[15:0] (66560&0xffff)
    g1[7] = 1;                                           // s1[47:16] (66560>>16)
    g2[0] = 4; g2[1] = 0; g2[2] = 0; g2[3] = 0;          // tensor_dim2=4
    g3[0] = 0; g3[1] = 0; g3[2] = 0; g3[3] = 0;
    g4[0] = 0; g4[1] = 0; g4[2] = 0; g4[3] = 0;          // VADDR4: unused
    g4[4] = 0; g4[5] = 0; g4[6] = 0; g4[7] = 0;
    __builtin_amdgcn_tensor_load_to_lds(g0, g1, g2, g3, g4, 0);
}

// ---------------------------------------------------------------------------
// Kernel 2: implicit GEMM, D = W * Y  via v_wmma_f32_16x16x32_bf16.
// M = Cout(512), N = spatial(8192), K = Cin*9(4608), 144 K-steps of 32.
// WG: 8 waves (4x2), tile 128x128; wave tile 32x64 = 2x4 WMMA accumulators.
// B staged per K-step by TDM into double-buffered LDS; A direct from global.
// ---------------------------------------------------------------------------
__global__ __launch_bounds__(256) void conv_wmma(const unsigned short* __restrict__ ypk,
                                                 const unsigned short* __restrict__ wpk,
                                                 const float* __restrict__ bias,
                                                 float* __restrict__ out) {
    __shared__ alignas(32) unsigned short smem[2 * 4096];   // 2 x 8KB B tiles

    const unsigned bx     = blockIdx.x;                  // 256 blocks: 4(M) x 64(N)
    const unsigned m_base = (bx & 3u) * 128u;
    const unsigned n_base = (bx >> 2) * 128u;            // 128 spatial = 4 ho x 32 wo
    const unsigned tid = threadIdx.x;
    const unsigned wv  = tid >> 5;
    const unsigned l   = tid & 31u;
    const unsigned g   = l >> 4, lm = l & 15u;
    const unsigned wm  = wv & 3u, wn = wv >> 2;          // 4 x 2 wave grid
    const unsigned cot0 = (bx & 3u) * 8u + wm * 2u;

    // Uniform tile origin for the TDM descriptor (all 128 cols in one image).
    const unsigned nimg0 = n_base >> 10;
    const unsigned ho0   = (n_base & 1023u) >> 5;
    const unsigned lds_base = (unsigned)(unsigned long long)(void*)&smem[0];

    // Per-lane output decode for the four B n-subtiles.
    unsigned nimg[4], ho[4], wo[4];
#pragma unroll
    for (int t = 0; t < 4; ++t) {
        unsigned s = n_base + wn * 64u + (unsigned)t * 16u + lm;
        nimg[t] = s >> 10; ho[t] = (s >> 5) & 31u; wo[t] = s & 31u;
    }

    v8f acc[2][4] = {};

    // Prologue: TDM B tile for K-step 0 into buffer 0.
    if (wv == 0u) {
        unsigned long long ga = (unsigned long long)(
            ((nimg0 * 65u + 2u * ho0) * 65u) << 9) * 2ull + (unsigned long long)(size_t)ypk;
        tdm_issue_b(lds_base, ga);
    }

    for (unsigned k = 0; k < 144u; ++k) {
        if (wv == 0u) __builtin_amdgcn_s_wait_tensorcnt(0);
        __syncthreads();                                  // buf[k&1] ready, prior reads done
        if (wv == 0u && k + 1u < 144u) {
            unsigned kn = k + 1u;
            unsigned kidx = kn >> 4, cb = kn & 15u;
            unsigned kh = kidx / 3u, kw = kidx % 3u;
            unsigned helem = ((nimg0 * 65u + 2u * ho0 + kh) * 65u + kw) * 512u + cb * 32u;
            tdm_issue_b(lds_base + ((kn & 1u) << 13),
                        (unsigned long long)helem * 2ull + (unsigned long long)(size_t)ypk);
        }

        const unsigned short* bbuf = smem + ((k & 1u) << 12);
        v16bf a0 = *(const v16bf*)(wpk + ((k * 32u + cot0      ) * 32u + l) * 16u);
        v16bf a1 = *(const v16bf*)(wpk + ((k * 32u + cot0 + 1u ) * 32u + l) * 16u);
#pragma unroll
        for (int t = 0; t < 4; ++t) {
            // col n_local = wn*64 + t*16 + lm; 32B at n_local*64 + g*32 bytes
            v16bf b = *(const v16bf*)(bbuf + ((wn * 64u + (unsigned)t * 16u + lm) << 5) + (g << 4));
            acc[0][t] = __builtin_amdgcn_wmma_f32_16x16x32_bf16(false, a0, false, b,
                          (short)0, acc[0][t], false, false);
            acc[1][t] = __builtin_amdgcn_wmma_f32_16x16x32_bf16(false, a1, false, b,
                          (short)0, acc[1][t], false, false);
        }
    }

    // Epilogue: C/D layout: M = r + 8*g, N = lane&15 (per 16x16 tile).
#pragma unroll
    for (int i = 0; i < 2; ++i) {
        unsigned cobase = m_base + wm * 32u + (unsigned)i * 16u + 8u * g;
#pragma unroll
        for (int r = 0; r < 8; ++r) {
            unsigned co = cobase + (unsigned)r;
            float bv = bias[co];
#pragma unroll
            for (int t = 0; t < 4; ++t) {
                unsigned o = ((nimg[t] * 512u + co) << 10) + (ho[t] << 5) + wo[t];
                out[o] = acc[i][t][r] + bv;
            }
        }
    }
}

// ---------------------------------------------------------------------------
extern "C" void kernel_launch(void* const* d_in, const int* in_sizes, int n_in,
                              void* d_out, int out_size, void* d_ws, size_t ws_size,
                              hipStream_t stream) {
    const float* x    = (const float*)d_in[0];   // [8,512,64,64]
    const float* wgt  = (const float*)d_in[1];   // [512,512,3,3]
    const float* bias = (const float*)d_in[2];   // [512]
    float* out = (float*)d_out;                  // [8,512,32,32]

    unsigned short* y   = (unsigned short*)d_ws;                       // 34,611,200 B
    unsigned short* wpk = (unsigned short*)((char*)d_ws + 34611200u);  // 4.72 MB

    pack_weights<<<9216, 256, 0, stream>>>(wgt, wpk);
    fir_nhwc<<<67600, 256, 0, stream>>>(x, y);
    conv_wmma<<<256, 256, 0, stream>>>(y, wpk, bias, out);
}
